// LSTMModel_13649406066853
// MI455X (gfx1250) — matrix operand order repaired and checked by reading, not compiled, not run on hardware
//
#include <hip/hip_runtime.h>

// ---------------- problem constants ----------------
#define TT 400
#define BB 256
#define HH 256
#define XEe 32
#define SEe 16
#define NXx 100
// DIN = 32 + 256 + 1 = 289, padded K for ih GEMM:
#define KP_IH 320          // 10 K-tiles of 32
#define N3H   768          // 3*H, 48 N-tiles of 16
#define KCAT  512          // 16 K-tiles
#define KIPW  320          // 304 real (288 zx + 16 demo)
#define KOF   320          // 305 real (288 max + 16 demo + 1 flag)

// d_out layout (floats): ipw[T*B] | f_out[B] | cf_out[B] | f_h[B*H]
#define OUT_FOUT  (TT*BB)
#define OUT_CFOUT (TT*BB + BB)
#define OUT_FH    (TT*BB + 2*BB)

// ---------------- types / helpers ----------------
typedef __attribute__((ext_vector_type(16))) __bf16 bf16x16;
typedef __attribute__((ext_vector_type(8)))  float  floatx8;

union Frag { bf16x16 v; uint4 q[2]; unsigned short u[16]; };

__device__ __forceinline__ unsigned short f2bf(float f) {
  union { float f; unsigned u; } x; x.f = f;
  unsigned u = x.u;
  u += 0x7FFFu + ((u >> 16) & 1u);            // round-to-nearest-even
  return (unsigned short)(u >> 16);
}
__device__ __forceinline__ float bf2f(unsigned short h) {
  union { unsigned u; float f; } x; x.u = ((unsigned)h) << 16;
  return x.f;
}
__device__ __forceinline__ floatx8 zero8() {
  floatx8 v;
#pragma unroll
  for (int i = 0; i < 8; ++i) v[i] = 0.f;
  return v;
}
__device__ __forceinline__ void unpack8(uint4 q, float* f) {
  union { uint4 q; unsigned short u[8]; } x; x.q = q;
#pragma unroll
  for (int i = 0; i < 8; ++i) f[i] = bf2f(x.u[i]);
}

// A fragment (16x32 bf16) from row-major bf16 in LDS, row stride ld (elements)
__device__ __forceinline__ bf16x16 load_a_lds(const unsigned short* z, int ld, int k0, int lane) {
  int m  = lane & 15;
  int kb = (lane >> 4) * 8;
  Frag f;
  f.q[0] = *(const uint4*)(z + m * ld + k0 + kb);        // K = kb..kb+7
  f.q[1] = *(const uint4*)(z + m * ld + k0 + kb + 16);   // K = kb+16..kb+23
  return f.v;
}
// A fragment from row-major bf16 in global memory (row base given in rows, ld in elements)
__device__ __forceinline__ bf16x16 load_a_gl(const unsigned short* a, size_t rowBase, int ld, int k0, int lane) {
  int m  = lane & 15;
  int kb = (lane >> 4) * 8;
  const unsigned short* p = a + (rowBase + m) * (size_t)ld + k0 + kb;
  Frag f;
  f.q[0] = *(const uint4*)p;
  f.q[1] = *(const uint4*)(p + 16);
  return f.v;
}
// B fragment (32x16 bf16) pre-swizzled: frag fi, lane-contiguous 16 elements
__device__ __forceinline__ bf16x16 load_b_frag(const unsigned short* w, size_t fi, int lane) {
  const unsigned short* p = w + fi * 512 + (size_t)lane * 16;
  Frag f;
  f.q[0] = *(const uint4*)p;
  f.q[1] = *(const uint4*)(p + 8);
  return f.v;
}
__device__ __forceinline__ floatx8 wmma_bf16(bf16x16 a, bf16x16 b, floatx8 c) {
  return __builtin_amdgcn_wmma_f32_16x16x32_bf16(false, a, false, b, (short)0, c, false, false);
}
__device__ __forceinline__ void store_c_lds(float* dst, int ld, int c0, int lane, floatx8 c) {
  int col = c0 + (lane & 15);
  int mb  = (lane >> 4) * 8;
#pragma unroll
  for (int r = 0; r < 8; ++r) dst[(mb + r) * ld + col] = c[r];
}

// ---------------- prep kernels ----------------
// fp32 [K][N] row-major -> bf16 WMMA B-fragment order, K zero-padded to Kpad
__global__ void k_swz(const float* __restrict__ src, unsigned short* __restrict__ dst,
                      int Kreal, int Kpad, int N) {
  int gl = blockIdx.x * blockDim.x + threadIdx.x;
  int ntiles = N >> 4;
  int F = (Kpad >> 5) * ntiles;
  if (gl >= F * 32) return;
  int fi = gl >> 5, lane = gl & 31;
  int kt = fi / ntiles, nt = fi - kt * ntiles;
  int n  = nt * 16 + (lane & 15);
  int kb = kt * 32 + ((lane >> 4) << 3);
#pragma unroll
  for (int e = 0; e < 16; ++e) {
    int k = kb + (e < 8 ? e : e + 8);
    float v = (k < Kreal) ? src[(size_t)k * N + n] : 0.f;
    dst[(size_t)gl * 16 + e] = f2bf(v);
  }
}

// x_emb[t][b][32] = x[b][t][:] @ w_x2emb + b_x2emb   (fp32)
__global__ void k_emb(const float* __restrict__ x, const float* __restrict__ w,
                      const float* __restrict__ bias, float* __restrict__ x_emb) {
  int t = blockIdx.x, b = threadIdx.x;
  float acc[XEe];
#pragma unroll
  for (int e = 0; e < XEe; ++e) acc[e] = bias[e];
  const float* xp = x + ((size_t)b * TT + t) * NXx;
  for (int f = 0; f < NXx; ++f) {
    float xv = xp[f];
#pragma unroll
    for (int e = 0; e < XEe; ++e) acc[e] += xv * w[f * XEe + e];
  }
  float* op = x_emb + ((size_t)t * BB + b) * XEe;
#pragma unroll
  for (int e = 0; e < XEe; ++e) op[e] = acc[e];
}

// demo[b][16], ftr[b] = sum_t x_fr[b][t]
__global__ void k_demo(const float* __restrict__ x_demo, const float* __restrict__ x_fr,
                       const float* __restrict__ w, const float* __restrict__ bias,
                       float* __restrict__ demo, float* __restrict__ ftr) {
  int b = threadIdx.x;
  float s = 0.f;
  for (int t = 0; t < TT; ++t) s += x_fr[(size_t)b * TT + t];
  ftr[b] = s;
  for (int e = 0; e < SEe; ++e) {
    float d = bias[e];
    for (int f = 0; f < 5; ++f) d += x_demo[b * 5 + f] * w[f * SEe + e];
    demo[b * SEe + e] = d;
  }
}

// ---------------- main persistent RNN kernel ----------------
// dynamic LDS byte offsets
constexpr int OFF_Z     = 0;        // ushort 16*320  (shared f/cf input row)
constexpr int OFF_HF    = 10240;    // ushort 16*256
constexpr int OFF_HCF   = 18432;    // ushort 16*256
constexpr int OFF_OLDF  = 26624;    // ushort 16*256
constexpr int OFF_OLDCF = 34816;    // ushort 16*256
constexpr int OFF_ZCAT  = 43008;    // ushort 16*512
constexpr int OFF_RZ    = 59392;    // float  16*512
constexpr int OFF_XN    = 92160;    // float  16*256
constexpr int OFF_HN    = 108544;   // float  16*256
constexpr int OFF_SC    = 124928;   // float  400*16
constexpr int OFF_RED   = 150528;   // float  512
constexpr int OFF_MAXF  = 152576;   // float  16*288
constexpr int OFF_MAXCF = 171008;   // float  16*288
constexpr int SMEM_MAIN = 189440;

__device__ __forceinline__ void build_z(
    int t, int gbBase, int tid, int isCf,
    unsigned short* sZ, const unsigned short* sOld, float* sMax,
    const float* x_emb, const float* x_fr, const float* ftr, const float* demo,
    unsigned short* ipw_in) {
  for (int idx = tid; idx < 16 * KP_IH; idx += 256) {
    int m = idx / KP_IH, c = idx - m * KP_IH;
    int gb = gbBase + m;
    float v;
    if (c < XEe)        v = x_emb[((size_t)t * BB + gb) * XEe + c];
    else if (c < 288)   v = bf2f(sOld[m * HH + (c - XEe)]);
    else if (c == 288)  v = isCf ? ((ftr[gb] > 0.f) ? 0.f : 1.f)
                               : x_fr[(size_t)gb * TT + t];
    else                v = 0.f;
    sZ[idx] = f2bf(v);
    if (c < 288) sMax[m * 288 + c] = fmaxf(sMax[m * 288 + c], v);
    if (!isCf) {
      unsigned short ip;
      if (c < 288)      ip = f2bf(v);
      else if (c < 304) ip = f2bf(demo[gb * SEe + (c - 288)]);
      else              ip = 0;
      ipw_in[((size_t)t * BB + gb) * KIPW + c] = ip;
    }
  }
  __syncthreads();
}

__device__ __forceinline__ void gru_step(
    int t, int gbBase, int tid,
    const unsigned short* sZ, unsigned short* sH, unsigned short* sOld,
    unsigned short* sZcat, float* sRZ, float* sXN, float* sHN,
    float* sSc, float* sRed,
    const unsigned short* wih_fr, const unsigned short* whh_fr, const unsigned short* wcat_fr,
    const float* bih, const float* bhh, const float* bcat,
    unsigned short* buf) {
  int wave = tid >> 5, lane = tid & 31;

  // ---- GEMM: gx = z@Wih (10 K-tiles), gh = h@Whh (8 K-tiles), 48 N-tiles ----
  for (int nti = 0; nti < 6; ++nti) {
    int nt = wave * 6 + nti;
    int c0 = nt * 16;
    if (c0 < 2 * HH) {  // r,z gates: combine gx+gh
      floatx8 acc = zero8();
#pragma unroll
      for (int kt = 0; kt < 10; ++kt)
        acc = wmma_bf16(load_a_lds(sZ, KP_IH, kt * 32, lane),
                        load_b_frag(wih_fr, (size_t)kt * 48 + nt, lane), acc);
#pragma unroll
      for (int kt = 0; kt < 8; ++kt)
        acc = wmma_bf16(load_a_lds(sH, HH, kt * 32, lane),
                        load_b_frag(whh_fr, (size_t)kt * 48 + nt, lane), acc);
      store_c_lds(sRZ, 2 * HH, c0, lane, acc);
    } else {            // n gate: keep gx, gh separate
      floatx8 ax = zero8();
#pragma unroll
      for (int kt = 0; kt < 10; ++kt)
        ax = wmma_bf16(load_a_lds(sZ, KP_IH, kt * 32, lane),
                       load_b_frag(wih_fr, (size_t)kt * 48 + nt, lane), ax);
      store_c_lds(sXN, HH, c0 - 2 * HH, lane, ax);
      floatx8 ah = zero8();
#pragma unroll
      for (int kt = 0; kt < 8; ++kt)
        ah = wmma_bf16(load_a_lds(sH, HH, kt * 32, lane),
                       load_b_frag(whh_fr, (size_t)kt * 48 + nt, lane), ah);
      store_c_lds(sHN, HH, c0 - 2 * HH, lane, ah);
    }
  }
  __syncthreads();

  // ---- gate math, update h, seed zcat ----
  for (int idx = tid; idx < 16 * HH; idx += 256) {
    int m = idx >> 8, j = idx & 255;
    float xr = sRZ[m * 512 + j]       + bih[j]        + bhh[j];
    float xz = sRZ[m * 512 + 256 + j] + bih[256 + j]  + bhh[256 + j];
    float xn = sXN[m * 256 + j]       + bih[512 + j];
    float hn = sHN[m * 256 + j]       + bhh[512 + j];
    float r = 1.f / (1.f + __expf(-xr));
    float z = 1.f / (1.f + __expf(-xz));
    float n = tanhf(xn + r * hn);
    float hnew = (1.f - z) * n + z * bf2f(sH[m * 256 + j]);
    unsigned short hb = f2bf(hnew);
    sH[m * 256 + j] = hb;
    sZcat[m * 512 + j] = hb;
    if (t == 0) sZcat[m * 512 + 256 + j] = hb;  // ctx = h at step 0
    buf[((size_t)t * BB + gbBase + m) * HH + j] = hb;
  }
  __syncthreads();

  // ---- attention over history (t > 0) ----
  if (t > 0) {
    int b = tid & 15, ts = tid >> 4;
    // scores
    const unsigned short* hp = sH + b * 256;
    for (int tt = ts; tt < t; tt += 16) {
      const unsigned short* bp = buf + ((size_t)tt * BB + gbBase + b) * HH;
      float acc = 0.f;
      for (int k8 = 0; k8 < 32; ++k8) {
        float fa[8], fb[8];
        unpack8(*(const uint4*)(hp + k8 * 8), fa);
        unpack8(*(const uint4*)(bp + k8 * 8), fb);
#pragma unroll
        for (int i = 0; i < 8; ++i) acc += fa[i] * fb[i];
      }
      sSc[tt * 16 + b] = acc;
    }
    __syncthreads();
    // softmax: max
    float lmax = -1e30f;
    for (int tt = ts; tt < t; tt += 16) lmax = fmaxf(lmax, sSc[tt * 16 + b]);
    sRed[ts * 16 + b] = lmax;
    __syncthreads();
    if (tid < 16) {
      float mx = -1e30f;
      for (int s = 0; s < 16; ++s) mx = fmaxf(mx, sRed[s * 16 + tid]);
      sRed[256 + tid] = mx;
    }
    __syncthreads();
    // softmax: exp + sum
    float mx = sRed[256 + b];
    float lsum = 0.f;
    for (int tt = ts; tt < t; tt += 16) {
      float e = __expf(sSc[tt * 16 + b] - mx);
      sSc[tt * 16 + b] = e;
      lsum += e;
    }
    sRed[ts * 16 + b] = lsum;
    __syncthreads();
    if (tid < 16) {
      float s0 = 0.f;
      for (int s = 0; s < 16; ++s) s0 += sRed[s * 16 + tid];
      sRed[272 + tid] = 1.f / s0;
    }
    __syncthreads();
    // weighted context -> zcat cols 256..511
    {
      int k0 = ts * 16;
      float acc[16];
#pragma unroll
      for (int i = 0; i < 16; ++i) acc[i] = 0.f;
      for (int tt = 0; tt < t; ++tt) {
        float w = sSc[tt * 16 + b];
        const unsigned short* bp = buf + ((size_t)tt * BB + gbBase + b) * HH + k0;
        float fb[8];
        unpack8(*(const uint4*)bp, fb);
#pragma unroll
        for (int i = 0; i < 8; ++i) acc[i] += w * fb[i];
        unpack8(*(const uint4*)(bp + 8), fb);
#pragma unroll
        for (int i = 0; i < 8; ++i) acc[8 + i] += w * fb[i];
      }
      float inv = sRed[272 + b];
#pragma unroll
      for (int i = 0; i < 16; ++i)
        sZcat[b * 512 + 256 + k0 + i] = f2bf(acc[i] * inv);
    }
    __syncthreads();
  }

  // ---- cat GEMM: old = tanh([h|ctx] @ Wcat + b) ----
  for (int nti = 0; nti < 2; ++nti) {
    int nt = wave * 2 + nti, c0 = nt * 16;
    floatx8 acc = zero8();
#pragma unroll
    for (int kt = 0; kt < 16; ++kt)
      acc = wmma_bf16(load_a_lds(sZcat, KCAT, kt * 32, lane),
                      load_b_frag(wcat_fr, (size_t)kt * 16 + nt, lane), acc);
    int col = c0 + (lane & 15), mb = (lane >> 4) * 8;
#pragma unroll
    for (int r = 0; r < 8; ++r)
      sOld[(mb + r) * HH + col] = f2bf(tanhf(acc[r] + bcat[col]));
  }
  __syncthreads();
}

__global__ __launch_bounds__(256) void k_main(
    const float* __restrict__ x_emb, const float* __restrict__ x_fr,
    const float* __restrict__ f_hx0, const float* __restrict__ cf_hx0,
    const float* __restrict__ demo, const float* __restrict__ ftr,
    const unsigned short* __restrict__ wihf, const unsigned short* __restrict__ whhf,
    const unsigned short* __restrict__ wihcf, const unsigned short* __restrict__ whhcf,
    const unsigned short* __restrict__ wcatf, const unsigned short* __restrict__ wcatcf,
    const float* __restrict__ bihf, const float* __restrict__ bhhf,
    const float* __restrict__ bihcf, const float* __restrict__ bhhcf,
    const float* __restrict__ bcatf, const float* __restrict__ bcatcf,
    unsigned short* __restrict__ fbuf, unsigned short* __restrict__ cfbuf,
    unsigned short* __restrict__ ipw_in,
    unsigned short* __restrict__ of_in, unsigned short* __restrict__ ocf_in) {
  extern __shared__ char smem[];
  unsigned short* sZ     = (unsigned short*)(smem + OFF_Z);
  unsigned short* sHf    = (unsigned short*)(smem + OFF_HF);
  unsigned short* sHcf   = (unsigned short*)(smem + OFF_HCF);
  unsigned short* sOldf  = (unsigned short*)(smem + OFF_OLDF);
  unsigned short* sOldcf = (unsigned short*)(smem + OFF_OLDCF);
  unsigned short* sZcat  = (unsigned short*)(smem + OFF_ZCAT);
  float* sRZ    = (float*)(smem + OFF_RZ);
  float* sXN    = (float*)(smem + OFF_XN);
  float* sHN    = (float*)(smem + OFF_HN);
  float* sSc    = (float*)(smem + OFF_SC);
  float* sRed   = (float*)(smem + OFF_RED);
  float* sMaxF  = (float*)(smem + OFF_MAXF);
  float* sMaxCf = (float*)(smem + OFF_MAXCF);

  int tid = threadIdx.x;
  int gbBase = blockIdx.x * 16;

  for (int idx = tid; idx < 16 * HH; idx += 256) {
    int m = idx >> 8, k = idx & 255;
    int gb = gbBase + m;
    unsigned short hf = f2bf(f_hx0[(size_t)gb * HH + k]);
    unsigned short hc = f2bf(cf_hx0[(size_t)gb * HH + k]);
    sHf[idx] = hf; sOldf[idx] = hf;
    sHcf[idx] = hc; sOldcf[idx] = hc;
  }
  for (int idx = tid; idx < 16 * 288; idx += 256) { sMaxF[idx] = -1e30f; sMaxCf[idx] = -1e30f; }
  __syncthreads();

  for (int t = 0; t < TT; ++t) {
    build_z(t, gbBase, tid, 0, sZ, sOldf, sMaxF, x_emb, x_fr, ftr, demo, ipw_in);
    gru_step(t, gbBase, tid, sZ, sHf, sOldf, sZcat, sRZ, sXN, sHN, sSc, sRed,
             wihf, whhf, wcatf, bihf, bhhf, bcatf, fbuf);
    build_z(t, gbBase, tid, 1, sZ, sOldcf, sMaxCf, x_emb, x_fr, ftr, demo, ipw_in);
    gru_step(t, gbBase, tid, sZ, sHcf, sOldcf, sZcat, sRZ, sXN, sHN, sSc, sRed,
             wihcf, whhcf, wcatcf, bihcf, bhhcf, bcatcf, cfbuf);
  }

  // epilogue: head-input rows [max | demo | flag | pad]
  for (int idx = tid; idx < 16 * KOF; idx += 256) {
    int m = idx / KOF, c = idx - m * KOF;
    int gb = gbBase + m;
    float ft = (ftr[gb] > 0.f) ? 1.f : 0.f;
    float vo, vc;
    if (c < 288)      { vo = sMaxF[m * 288 + c]; vc = sMaxCf[m * 288 + c]; }
    else if (c < 304) { vo = vc = demo[gb * SEe + (c - 288)]; }
    else if (c == 304){ vo = ft; vc = 1.f - ft; }
    else              { vo = vc = 0.f; }
    of_in[(size_t)gb * KOF + c]  = f2bf(vo);
    ocf_in[(size_t)gb * KOF + c] = f2bf(vc);
  }
}

// ---------------- ipw head: relu(in @ W + b) @ wout ----------------
__global__ __launch_bounds__(256) void k_ipw(
    const unsigned short* __restrict__ ipw_in, const unsigned short* __restrict__ wfr,
    const float* __restrict__ bias, const float* __restrict__ wout,
    float* __restrict__ out) {
  extern __shared__ char smem[];
  float* sH = (float*)smem;                       // 64*256 fp32
  int tid = threadIdx.x, wave = tid >> 5, lane = tid & 31;
  size_t r0 = (size_t)blockIdx.x * 64;
  int mt = wave & 3, nh = wave >> 2;
  size_t rowBase = r0 + (size_t)mt * 16;
  for (int nti = 0; nti < 8; ++nti) {
    int nt = nh * 8 + nti, c0 = nt * 16;
    floatx8 acc = zero8();
#pragma unroll
    for (int kt = 0; kt < 10; ++kt)
      acc = wmma_bf16(load_a_gl(ipw_in, rowBase, KIPW, kt * 32, lane),
                      load_b_frag(wfr, (size_t)kt * 16 + nt, lane), acc);
    int col = c0 + (lane & 15), mb = (lane >> 4) * 8;
#pragma unroll
    for (int r = 0; r < 8; ++r)
      sH[(mt * 16 + mb + r) * HH + col] = fmaxf(acc[r] + bias[col], 0.f);
  }
  __syncthreads();
  if (tid < 64) {
    float a = 0.f;
    for (int k = 0; k < HH; ++k) a += sH[tid * HH + k] * wout[k];
    out[r0 + tid] = a;
  }
}

// ---------------- final heads (f / cf) ----------------
__global__ __launch_bounds__(256) void k_head(
    const unsigned short* __restrict__ of_in, const unsigned short* __restrict__ ocf_in,
    const unsigned short* __restrict__ wof_fr, const unsigned short* __restrict__ wocf_fr,
    const float* __restrict__ b_of, const float* __restrict__ b_ocf,
    const float* __restrict__ w_ho_f, const float* __restrict__ w_ho_cf,
    float* __restrict__ d_out) {
  __shared__ float sH[16 * HH];
  int by = blockIdx.y;
  const unsigned short* in  = by ? ocf_in : of_in;
  const unsigned short* wfr = by ? wocf_fr : wof_fr;
  const float* bias = by ? b_ocf : b_of;
  int tid = threadIdx.x, wave = tid >> 5, lane = tid & 31;
  size_t r0 = (size_t)blockIdx.x * 16;
  for (int nti = 0; nti < 2; ++nti) {
    int nt = wave * 2 + nti, c0 = nt * 16;
    floatx8 acc = zero8();
#pragma unroll
    for (int kt = 0; kt < 10; ++kt)
      acc = wmma_bf16(load_a_gl(in, r0, KOF, kt * 32, lane),
                      load_b_frag(wfr, (size_t)kt * 16 + nt, lane), acc);
    int col = c0 + (lane & 15), mb = (lane >> 4) * 8;
#pragma unroll
    for (int r = 0; r < 8; ++r)
      sH[(mb + r) * HH + col] = fmaxf(acc[r] + bias[col], 0.f);
  }
  __syncthreads();
  if (by == 0) {
    for (int idx = tid; idx < 16 * HH; idx += 256)
      d_out[OUT_FH + (r0 + (idx >> 8)) * HH + (idx & 255)] = sH[idx];
    if (tid < 16) {
      float a = 0.f;
      for (int k = 0; k < HH; ++k) a += sH[tid * HH + k] * w_ho_f[k];
      d_out[OUT_FOUT + r0 + tid] = a;
    }
  } else {
    if (tid < 16) {
      float a = 0.f;
      for (int k = 0; k < HH; ++k) a += sH[tid * HH + k] * w_ho_cf[k];
      d_out[OUT_CFOUT + r0 + tid] = a;
    }
  }
}

// ---------------- host launcher ----------------
extern "C" void kernel_launch(void* const* d_in, const int* in_sizes, int n_in,
                              void* d_out, int out_size, void* d_ws, size_t ws_size,
                              hipStream_t stream) {
  (void)in_sizes; (void)n_in; (void)out_size; (void)ws_size;
  const float* x        = (const float*)d_in[0];
  const float* x_demo   = (const float*)d_in[1];
  const float* x_fr     = (const float*)d_in[2];
  const float* f_hx0    = (const float*)d_in[3];
  const float* cf_hx0   = (const float*)d_in[4];
  const float* w_x2emb  = (const float*)d_in[5];
  const float* b_x2emb  = (const float*)d_in[6];
  const float* w_static = (const float*)d_in[7];
  const float* b_static = (const float*)d_in[8];
  const float* wih_f    = (const float*)d_in[9];
  const float* whh_f    = (const float*)d_in[10];
  const float* bih_f    = (const float*)d_in[11];
  const float* bhh_f    = (const float*)d_in[12];
  const float* wih_cf   = (const float*)d_in[13];
  const float* whh_cf   = (const float*)d_in[14];
  const float* bih_cf   = (const float*)d_in[15];
  const float* bhh_cf   = (const float*)d_in[16];
  const float* w_cat_f  = (const float*)d_in[17];
  const float* b_cat_f  = (const float*)d_in[18];
  const float* w_cat_cf = (const float*)d_in[19];
  const float* b_cat_cf = (const float*)d_in[20];
  const float* w_ipw    = (const float*)d_in[21];
  const float* b_ipw    = (const float*)d_in[22];
  const float* w_ipw_out= (const float*)d_in[23];
  const float* w_of     = (const float*)d_in[24];
  const float* b_of     = (const float*)d_in[25];
  const float* w_ho_f   = (const float*)d_in[26];
  const float* w_ocf    = (const float*)d_in[27];
  const float* b_ocf    = (const float*)d_in[28];
  const float* w_ho_cf  = (const float*)d_in[29];

  size_t off = 0;
  auto carve = [&](size_t bytes) -> char* {
    off = (off + 255) & ~(size_t)255;
    char* p = (char*)d_ws + off;
    off += bytes;
    return p;
  };
  unsigned short* wihf_fr  = (unsigned short*)carve((size_t)10 * 48 * 512 * 2);
  unsigned short* whhf_fr  = (unsigned short*)carve((size_t)8  * 48 * 512 * 2);
  unsigned short* wihcf_fr = (unsigned short*)carve((size_t)10 * 48 * 512 * 2);
  unsigned short* whhcf_fr = (unsigned short*)carve((size_t)8  * 48 * 512 * 2);
  unsigned short* wcatf_fr = (unsigned short*)carve((size_t)16 * 16 * 512 * 2);
  unsigned short* wcatcf_fr= (unsigned short*)carve((size_t)16 * 16 * 512 * 2);
  unsigned short* wipw_fr  = (unsigned short*)carve((size_t)10 * 16 * 512 * 2);
  unsigned short* wof_fr   = (unsigned short*)carve((size_t)10 * 16 * 512 * 2);
  unsigned short* wocf_fr  = (unsigned short*)carve((size_t)10 * 16 * 512 * 2);
  float* x_emb   = (float*)carve((size_t)TT * BB * XEe * 4);
  float* demo    = (float*)carve((size_t)BB * SEe * 4);
  float* ftr     = (float*)carve((size_t)BB * 4);
  unsigned short* fbuf   = (unsigned short*)carve((size_t)TT * BB * HH * 2);
  unsigned short* cfbuf  = (unsigned short*)carve((size_t)TT * BB * HH * 2);
  unsigned short* ipw_in = (unsigned short*)carve((size_t)TT * BB * KIPW * 2);
  unsigned short* of_in  = (unsigned short*)carve((size_t)BB * KOF * 2);
  unsigned short* ocf_in = (unsigned short*)carve((size_t)BB * KOF * 2);

  (void)hipFuncSetAttribute((const void*)k_main, hipFuncAttributeMaxDynamicSharedMemorySize, SMEM_MAIN);
  (void)hipFuncSetAttribute((const void*)k_ipw,  hipFuncAttributeMaxDynamicSharedMemorySize, 64 * HH * 4);

  auto swz = [&](const float* src, unsigned short* dst, int Kreal, int Kpad, int N) {
    int total = (Kpad / 32) * (N / 16) * 32;
    k_swz<<<(total + 255) / 256, 256, 0, stream>>>(src, dst, Kreal, Kpad, N);
  };
  swz(wih_f,   wihf_fr,  289, KP_IH, N3H);
  swz(whh_f,   whhf_fr,  256, 256,   N3H);
  swz(wih_cf,  wihcf_fr, 289, KP_IH, N3H);
  swz(whh_cf,  whhcf_fr, 256, 256,   N3H);
  swz(w_cat_f, wcatf_fr, 512, KCAT,  HH);
  swz(w_cat_cf,wcatcf_fr,512, KCAT,  HH);
  swz(w_ipw,   wipw_fr,  304, KIPW,  HH);
  swz(w_of,    wof_fr,   305, KOF,   HH);
  swz(w_ocf,   wocf_fr,  305, KOF,   HH);

  k_emb<<<TT, BB, 0, stream>>>(x, w_x2emb, b_x2emb, x_emb);
  k_demo<<<1, BB, 0, stream>>>(x_demo, x_fr, w_static, b_static, demo, ftr);

  k_main<<<BB / 16, 256, SMEM_MAIN, stream>>>(
      x_emb, x_fr, f_hx0, cf_hx0, demo, ftr,
      wihf_fr, whhf_fr, wihcf_fr, whhcf_fr, wcatf_fr, wcatcf_fr,
      bih_f, bhh_f, bih_cf, bhh_cf, b_cat_f, b_cat_cf,
      fbuf, cfbuf, ipw_in, of_in, ocf_in);

  k_ipw<<<(TT * BB) / 64, 256, 64 * HH * 4, stream>>>(
      ipw_in, wipw_fr, b_ipw, w_ipw_out, (float*)d_out);

  k_head<<<dim3(BB / 16, 2), 256, 0, stream>>>(
      of_in, ocf_in, wof_fr, wocf_fr, b_of, b_ocf, w_ho_f, w_ho_cf, (float*)d_out);
}
